// Attention_19164144075349
// MI455X (gfx1250) — matrix operand order repaired and checked by reading, not compiled, not run on hardware
//
#include <hip/hip_runtime.h>
#include <math.h>

// Problem constants (match the reference)
#define BB 32
#define OO 64
#define MM 128
#define HH 4
#define EE 64
#define HE (HH * EE) /* 256 */

typedef __attribute__((ext_vector_type(16))) __bf16 v16bf;
typedef __attribute__((ext_vector_type(8)))  float  v8f;

static __device__ __forceinline__ __bf16 tobf(float f) { return (__bf16)f; }

static __device__ __forceinline__ v8f wmma_bf16(v16bf a, v16bf b, v8f c) {
  return __builtin_amdgcn_wmma_f32_16x16x32_bf16(false, a, false, b, (short)0,
                                                 c, false, false);
}

// ---------------------------------------------------------------------------
// WMMA tile loaders for v_wmma_f32_16x16x32_bf16 (wave32 layouts, CDNA5 ISA
// 7.12.2).  A is 16x32 (row m = lane&15; lane-half selects K sub-blocks
// 0..15 vs 16..31, same K order as B).  B is 32x16 (col n = lane&15; lane-half
// selects K 0..15 vs 16..31).  C/D 16x16: n = lane&15, m = j + 8*(lane>>4).
// LDA/LDB are compile-time so strided element addresses fold to immediate
// offsets off a single 64-bit base.
// ---------------------------------------------------------------------------
template <int LDA>
static __device__ __forceinline__ v16bf load_a_tile(const float* src, int row0,
                                                    int k0) {
  const int lane = threadIdx.x & 31;
  const int hlf  = lane >> 4;
  const float* p = src + (row0 + (lane & 15)) * LDA + k0 + hlf * 16;
  v16bf a;
#pragma unroll
  for (int j = 0; j < 8; ++j) a[j] = tobf(p[j]);          // K = kbase + 0..7
#pragma unroll
  for (int j = 0; j < 8; ++j) a[8 + j] = tobf(p[8 + j]);  // K = kbase + 8..15
  return a;
}

template <int LDB>
static __device__ __forceinline__ v16bf load_b_tile(const float* src, int k0,
                                                    int col0) {
  const int lane = threadIdx.x & 31;
  const int hlf  = lane >> 4;
  const float* p = src + (k0 + hlf * 16) * LDB + col0 + (lane & 15);
  v16bf b;
#pragma unroll
  for (int j = 0; j < 16; ++j) b[j] = tobf(p[j * LDB]);
  return b;
}

// ---------------------------------------------------------------------------
// Y = X*W + bias : (rows x K) * (K x N).  One 16x32 output slab per wave
// (two adjacent 16x16 N-tiles sharing one A tile).  Grid must supply exactly
// (rows/16)*(N/32)/8 blocks of 256 threads.  K, N compile-time: K-loop fully
// unrolls, B addressing folds to immediate offsets.
// ---------------------------------------------------------------------------
template <int K, int N>
__global__ void gemm_bias_kernel(const float* __restrict__ X,
                                 const float* __restrict__ W,
                                 const float* __restrict__ bias,
                                 float* __restrict__ Y) {
  constexpr int NT = N / 32;  // column-pair tiles per row-tile
  const int gw   = blockIdx.x * (blockDim.x >> 5) + (threadIdx.x >> 5);
  const int row0 = (gw / NT) * 16;
  const int col0 = (gw % NT) * 32;

  v8f acc0 = {}, acc1 = {};
#pragma unroll
  for (int k0 = 0; k0 < K; k0 += 32) {
    if (k0 + 32 < K) {  // folds at compile time (loop fully unrolled)
      __builtin_prefetch(X + row0 * K + k0 + 32, 0, 1);
      __builtin_prefetch(W + (k0 + 32) * N + col0, 0, 1);
    }
    v16bf a  = load_a_tile<K>(X, row0, k0);
    v16bf b0 = load_b_tile<N>(W, k0, col0);
    v16bf b1 = load_b_tile<N>(W, k0, col0 + 16);
    acc0 = wmma_bf16(a, b0, acc0);
    acc1 = wmma_bf16(a, b1, acc1);
  }

  const int lane = threadIdx.x & 31;
  const int hlf  = lane >> 4;
  const int n0   = col0 + (lane & 15);
  const int n1   = n0 + 16;
  const float bv0 = bias[n0];
  const float bv1 = bias[n1];
#pragma unroll
  for (int j = 0; j < 8; ++j) {
    const int m = row0 + hlf * 8 + j;
    Y[m * N + n0] = acc0[j] + bv0;
    Y[m * N + n1] = acc1[j] + bv1;
  }
}

// ---------------------------------------------------------------------------
// Logits + softmax.  One block per batch b.  Context [128 x 64] staged in LDS
// with stride 65 (64 banks -> conflict-free column access).  Each wave owns 32
// (o,h) pairs; each lane owns 4 memory rows (m = lane + 32c) so the e-loop is
// a pure tanh+FMA stream (TRANS32 co-executes with VALU) and the softmax needs
// only one wave-max and one wave-sum.
// ---------------------------------------------------------------------------
__global__ void logits_softmax_kernel(const float* __restrict__ ctx,
                                      const float* __restrict__ qh,
                                      const float* __restrict__ w_logit,
                                      const float* __restrict__ b_logit,
                                      const float* __restrict__ temp,
                                      float* __restrict__ probs) {
  __shared__ float sctx[MM][EE + 1];
  __shared__ float sw[EE];

  const int b   = blockIdx.x;
  const int tid = threadIdx.x;
  for (int i = tid; i < MM * EE; i += blockDim.x) {
    sctx[i >> 6][i & 63] = ctx[b * MM * EE + i];
  }
  if (tid < EE) sw[tid] = w_logit[tid];
  __syncthreads();

  const float b0   = b_logit[0];
  const float invT = 1.0f / temp[0];
  const int lane = tid & 31;
  const int w    = tid >> 5;

  for (int pp = 0; pp < (OO * HH) / 8; ++pp) {  // 32 pairs per wave
    const int pair = w * 32 + pp;
    const int o = pair >> 2;
    const int h = pair & 3;
    const float* qrow = qh + (b * OO + o) * HE + h * EE;  // wave-uniform

    float a0 = 0.f, a1 = 0.f, a2 = 0.f, a3 = 0.f;
    for (int e = 0; e < EE; ++e) {
      const float qv = qrow[e];
      const float wv = sw[e];
      a0 += tanhf(sctx[lane      ][e] + qv) * wv;
      a1 += tanhf(sctx[lane + 32 ][e] + qv) * wv;
      a2 += tanhf(sctx[lane + 64 ][e] + qv) * wv;
      a3 += tanhf(sctx[lane + 96 ][e] + qv) * wv;
    }
    float l0 = (a0 + b0) * invT, l1 = (a1 + b0) * invT;
    float l2 = (a2 + b0) * invT, l3 = (a3 + b0) * invT;

    float mx = fmaxf(fmaxf(l0, l1), fmaxf(l2, l3));
#pragma unroll
    for (int off = 16; off; off >>= 1) mx = fmaxf(mx, __shfl_xor(mx, off, 32));

    const float e0 = __expf(l0 - mx), e1 = __expf(l1 - mx);
    const float e2 = __expf(l2 - mx), e3 = __expf(l3 - mx);
    float s = e0 + e1 + e2 + e3;
#pragma unroll
    for (int off = 16; off; off >>= 1) s += __shfl_xor(s, off, 32);
    const float inv = 1.0f / s;

    float* pr = probs + ((b * HH + h) * OO + o) * MM;  // [B][H][O][M]
    pr[lane]      = e0 * inv;
    pr[lane + 32] = e1 * inv;
    pr[lane + 64] = e2 * inv;
    pr[lane + 96] = e3 * inv;
  }
}

// ---------------------------------------------------------------------------
// heads = leaky_relu(probs @ memory).  One block per (b,h): GEMM 64x128x64.
// Wave w owns o-tile (w>>1) and e-tile pair 2*(w&1) -> one A tile feeds two
// WMMAs per K step.  Output stored as [B][O][H*E] for the final projection.
// ---------------------------------------------------------------------------
__global__ void attend_kernel(const float* __restrict__ probs,
                              const float* __restrict__ mem,
                              float* __restrict__ heads) {
  const int b = blockIdx.x >> 2;
  const int h = blockIdx.x & 3;
  const int w = threadIdx.x >> 5;
  const float* A  = probs + (b * HH + h) * OO * MM;  // [64][128]
  const float* Bm = mem + b * MM * EE;               // [128][64]

  const int row0 = (w >> 1) * 16;      // o tile
  const int col0 = (w & 1) * 32;       // e tile pair
  v8f acc0 = {}, acc1 = {};
#pragma unroll
  for (int k0 = 0; k0 < MM; k0 += 32) {
    v16bf av  = load_a_tile<MM>(A, row0, k0);
    v16bf bv0 = load_b_tile<EE>(Bm, k0, col0);
    v16bf bv1 = load_b_tile<EE>(Bm, k0, col0 + 16);
    acc0 = wmma_bf16(av, bv0, acc0);
    acc1 = wmma_bf16(av, bv1, acc1);
  }

  const int lane = threadIdx.x & 31;
  const int hlf  = lane >> 4;
  const int e0c  = col0 + (lane & 15);
#pragma unroll
  for (int j = 0; j < 8; ++j) {
    const int o = row0 + hlf * 8 + j;
    float v0 = acc0[j];
    float v1 = acc1[j];
    v0 = v0 > 0.0f ? v0 : 0.01f * v0;  // leaky relu
    v1 = v1 > 0.0f ? v1 : 0.01f * v1;
    heads[(b * OO + o) * HE + h * EE + e0c]      = v0;
    heads[(b * OO + o) * HE + h * EE + e0c + 16] = v1;
  }
}

// ---------------------------------------------------------------------------
extern "C" void kernel_launch(void* const* d_in, const int* in_sizes, int n_in,
                              void* d_out, int out_size, void* d_ws,
                              size_t ws_size, hipStream_t stream) {
  const float* query   = (const float*)d_in[0];
  const float* context = (const float*)d_in[1];
  const float* memory  = (const float*)d_in[2];
  const float* W_ch    = (const float*)d_in[3];
  const float* b_ch    = (const float*)d_in[4];
  const float* w_logit = (const float*)d_in[5];
  const float* b_logit = (const float*)d_in[6];
  const float* W_rh    = (const float*)d_in[7];
  const float* b_rh    = (const float*)d_in[8];
  const float* temp    = (const float*)d_in[9];
  float* out = (float*)d_out;

  // Workspace partition (all L2-resident: ~8 MB total)
  float* qws   = (float*)d_ws;                 // [2048][256]
  float* probs = qws + (BB * OO) * HE;         // [B][H][O][M] = 1M floats
  float* heads = probs + BB * HH * OO * MM;    // [2048][256]

  // 1) q = query @ W_ch + b_ch : (2048 x 64) * (64 x 256)
  //    slabs = 128 * (256/32) = 1024 waves -> 128 blocks of 8 waves
  gemm_bias_kernel<EE, HE><<<128, 256, 0, stream>>>(query, W_ch, b_ch, qws);

  // 2) logits + softmax (tanh stage), one block per batch
  logits_softmax_kernel<<<BB, 256, 0, stream>>>(context, qws, w_logit, b_logit,
                                                temp, probs);

  // 3) heads = leaky_relu(probs @ memory), one block per (b,h)
  attend_kernel<<<BB * HH, 256, 0, stream>>>(probs, memory, heads);

  // 4) out = heads @ W_rh + b_rh : (2048 x 256) * (256 x 64)
  //    slabs = 128 * (64/32) = 256 waves -> 32 blocks of 8 waves
  gemm_bias_kernel<HE, EE><<<32, 256, 0, stream>>>(heads, W_rh, b_rh, out);
}
